// Gemma3MHAttention_86157043958050
// MI455X (gfx1250) — matrix operand order repaired and checked by reading, not compiled, not run on hardware
//
#include <hip/hip_runtime.h>

// ---------------------------------------------------------------------------
// Gemma3 sliding-window GQA block for MI455X (gfx1250, wave32, WMMA bf16).
// Pipeline: cvt(f32->bf16) -> QKV GEMM (WMMA + TDM-staged LDS tiles)
//           -> RMSNorm+RoPE (V transposed) -> flash attention (WMMA)
//           -> output GEMM (WMMA + TDM, f32 out)
// ---------------------------------------------------------------------------

typedef __bf16 bf16;
typedef __attribute__((ext_vector_type(16))) __bf16        v16bf;
typedef __attribute__((ext_vector_type(8)))  float         v8f;
typedef __attribute__((ext_vector_type(8)))  unsigned int  v8u;
typedef __attribute__((ext_vector_type(4)))  __bf16        v4bf;
typedef __attribute__((ext_vector_type(4)))  unsigned int  u32x4;
typedef __attribute__((ext_vector_type(8)))  int           i32x8;
typedef __attribute__((ext_vector_type(4)))  int           i32x4;

#define BN   2
#define TT   2048
#define EE   2048
#define NH   16
#define NKV  8
#define DHD  128
#define WND  512

__device__ __forceinline__ int imin(int a, int b) { return a < b ? a : b; }

__device__ __forceinline__ v8f zero8() {
    v8f z;
#pragma unroll
    for (int i = 0; i < 8; ++i) z[i] = 0.0f;
    return z;
}

// Load one 16x32 (A) or 32x16 (B) bf16 fragment: two 16-byte chunks per lane.
__device__ __forceinline__ v16bf frag16(const bf16* p0, const bf16* p1) {
    uint4 a = *(const uint4*)p0;
    uint4 b = *(const uint4*)p1;
    v8u u;
    u[0] = a.x; u[1] = a.y; u[2] = a.z; u[3] = a.w;
    u[4] = b.x; u[5] = b.y; u[6] = b.z; u[7] = b.w;
    return __builtin_bit_cast(v16bf, u);
}

__device__ __forceinline__ v8f wmma_bf16(v16bf a, v16bf b, v8f c) {
    return __builtin_amdgcn_wmma_f32_16x16x32_bf16(
        /*neg_a=*/false, a, /*neg_b=*/false, b,
        /*c_mod=*/(short)0, c, /*reuse_a=*/false, /*reuse_b=*/false);
}

// ---------------------------------------------------------------------------
// Tensor Data Mover: async 2D tile load (rows x cols bf16, row stride in
// elements) from global memory into LDS.  D# layout per CDNA5 ISA §8.3-8.5:
//   group0: [1:0]=count=1, [63:32]=lds_addr, [120:64]=global_addr, [127:126]=2
//   group1: [17:16]=data_size(1=2B), [79:48]=tensor_dim0, [111:80]=tensor_dim1,
//           [127:112]=tile_dim0, [143:128]=tile_dim1, [207:160]=dim0_stride
// Shared-aperture flat addresses are 4GiB aligned, so addr[31:0] is the LDS
// byte offset (ISA §10.2 LDS aperture truncation).
// ---------------------------------------------------------------------------
__device__ __forceinline__ void tdm_load_2d(void* lds, const void* g,
                                            unsigned rows, unsigned cols,
                                            unsigned stride_elems) {
    const unsigned lds_off = (unsigned)(uintptr_t)lds;
    const unsigned long long ga = (unsigned long long)(uintptr_t)g;
    u32x4 g0;
    g0[0] = 1u;                                   // count=1, load, user mode
    g0[1] = lds_off;                              // lds_addr
    g0[2] = (unsigned)ga;                         // global_addr[31:0]
    g0[3] = (unsigned)((ga >> 32) & 0x1FFFFFFu)   // global_addr[56:32]
          | (2u << 30);                           // type = 2 ("image")
    i32x8 g1;
    g1[0] = (int)(1u << 16);                      // data_size = 2 bytes
    g1[1] = (int)((cols & 0xFFFFu) << 16);        // tensor_dim0 lo16
    g1[2] = (int)((cols >> 16) | ((rows & 0xFFFFu) << 16));  // d0 hi | d1 lo
    g1[3] = (int)((rows >> 16) | ((cols & 0xFFFFu) << 16));  // d1 hi | tile_dim0
    g1[4] = (int)(rows & 0xFFFFu);                // tile_dim1 (tile_dim2 = 0)
    g1[5] = (int)stride_elems;                    // tensor_dim0_stride[31:0]
    g1[6] = 0;
    g1[7] = 0;
    i32x4 g2; g2[0] = 0; g2[1] = 0; g2[2] = 0; g2[3] = 0;
    i32x4 g3 = g2;
#if __clang_major__ >= 23
    i32x8 g4;
#pragma unroll
    for (int i = 0; i < 8; ++i) g4[i] = 0;
    __builtin_amdgcn_tensor_load_to_lds(g0, g1, g2, g3, g4, 0);
#else
    __builtin_amdgcn_tensor_load_to_lds(g0, g1, g2, g3, 0);
#endif
}

// ---------------------------------------------------------------------------
// Kernel 0: f32 -> bf16 conversion (vectorized, n divisible by 4)
// ---------------------------------------------------------------------------
__global__ __launch_bounds__(256) void cvt_f32_bf16(const float* __restrict__ s,
                                                    bf16* __restrict__ d, int n) {
    int i = (blockIdx.x * 256 + threadIdx.x) * 4;
    if (i < n) {
        float4 v = *(const float4*)(s + i);
        v4bf o;
        o[0] = (bf16)v.x; o[1] = (bf16)v.y; o[2] = (bf16)v.z; o[3] = (bf16)v.w;
        *(v4bf*)(d + i) = o;
    }
}

// ---------------------------------------------------------------------------
// Kernel 1/4: C[M][N] = A[M][K] * B[N][K]^T, bf16 in, f32 accum.
// Block tile 128x128, 8 waves as 4(M) x 2(N), wave tile 32x64 (2x4 WMMA tiles),
// K chunked by 32.  A/B tiles staged into double-buffered LDS by the Tensor
// Data Mover (wave 0 issues, TENSORcnt + barriers synchronize).
// ---------------------------------------------------------------------------
template <bool OUT_BF16>
__global__ __launch_bounds__(256) void gemm_bf16nt(const bf16* __restrict__ A,
                                                   const bf16* __restrict__ Bm,
                                                   void* __restrict__ Cout,
                                                   int M, int N, int K) {
    __shared__ __align__(16) bf16 lsA[2][128 * 32];
    __shared__ __align__(16) bf16 lsB[2][128 * 32];

    const int tid  = threadIdx.x;
    const int lane = tid & 31;
    const int wv   = tid >> 5;
    const int half = lane >> 4;
    const int col  = lane & 15;
    const int wm   = wv >> 1;   // 0..3 : M-wave
    const int wn   = wv & 1;    // 0..1 : N-wave
    const int m0 = blockIdx.y * 128;
    const int n0 = blockIdx.x * 128;

    const bf16* gA = A  + (size_t)m0 * K;
    const bf16* gB = Bm + (size_t)n0 * K;

    // Prologue: TDM-stage K-chunk 0 into buffer 0.
    if (wv == 0) {
        tdm_load_2d(&lsA[0][0], gA, 128, 32, (unsigned)K);
        tdm_load_2d(&lsB[0][0], gB, 128, 32, (unsigned)K);
    }

    v8f acc[2][4];
#pragma unroll
    for (int i = 0; i < 2; ++i)
#pragma unroll
        for (int j = 0; j < 4; ++j) acc[i][j] = zero8();

    const int KT = K / 32;
    for (int kt = 0; kt < KT; ++kt) {
        const int cur = kt & 1;
        __syncthreads();   // all waves done reading buffer cur^1 (prev iter)
        if (wv == 0) {
            if (kt + 1 < KT) {
                const int kc = (kt + 1) * 32;
                tdm_load_2d(&lsA[cur ^ 1][0], gA + kc, 128, 32, (unsigned)K);
                tdm_load_2d(&lsB[cur ^ 1][0], gB + kc, 128, 32, (unsigned)K);
                // 2 newest outstanding; <=2 left  => chunk `cur` complete.
                __builtin_amdgcn_s_wait_tensorcnt(2);
            } else {
                __builtin_amdgcn_s_wait_tensorcnt(0);
            }
        }
        __syncthreads();   // buffer cur ready for every wave

        v16bf af[2], bfr[4];
#pragma unroll
        for (int i = 0; i < 2; ++i) {
            const bf16* p = &lsA[cur][(wm * 32 + i * 16 + col) * 32 + half * 16];
            af[i] = frag16(p, p + 8);
        }
#pragma unroll
        for (int j = 0; j < 4; ++j) {
            const bf16* p = &lsB[cur][(wn * 64 + j * 16 + col) * 32 + half * 16];
            bfr[j] = frag16(p, p + 8);
        }
#pragma unroll
        for (int i = 0; i < 2; ++i)
#pragma unroll
            for (int j = 0; j < 4; ++j)
                acc[i][j] = wmma_bf16(af[i], bfr[j], acc[i][j]);
    }

#pragma unroll
    for (int i = 0; i < 2; ++i)
#pragma unroll
        for (int j = 0; j < 4; ++j) {
            const int mrow = m0 + wm * 32 + i * 16 + half * 8;
            const int ncol = n0 + wn * 64 + j * 16 + col;
            if (OUT_BF16) {
                bf16* C = (bf16*)Cout;
#pragma unroll
                for (int r = 0; r < 8; ++r)
                    C[(size_t)(mrow + r) * N + ncol] = (bf16)acc[i][j][r];
            } else {
                float* C = (float*)Cout;
#pragma unroll
                for (int r = 0; r < 8; ++r)
                    C[(size_t)(mrow + r) * N + ncol] = acc[i][j][r];
            }
        }
}

// ---------------------------------------------------------------------------
// Kernel 2: per-head RMSNorm + RoPE + q-scale. One wave per (token, slot).
// Slots: 0..15 = q heads, 16..23 = k heads, 24..31 = v heads (copy, transposed).
// V written as [B,HKV,DH,T] so attention B-fragments are contiguous loads.
// ---------------------------------------------------------------------------
__global__ __launch_bounds__(256) void rmsrope(const bf16* __restrict__ qkv,
                                               const float* __restrict__ qw,
                                               const float* __restrict__ kw,
                                               bf16* __restrict__ qb,
                                               bf16* __restrict__ kb,
                                               bf16* __restrict__ vT) {
    const int lane = threadIdx.x & 31;
    const int wid  = blockIdx.x * 8 + (threadIdx.x >> 5);
    const int slot = wid & 31;
    const int m    = wid >> 5;          // 0..B*T-1
    const int b    = m >> 11;
    const int t    = m & (TT - 1);

    int off;
    if (slot < 16)      off = slot * DHD;
    else if (slot < 24) off = 2048 + (slot - 16) * DHD;
    else                off = 3072 + (slot - 24) * DHD;

    const bf16* src = qkv + (size_t)m * 4096 + off;
    const int d0 = lane, d1 = lane + 32, d2 = lane + 64, d3 = lane + 96;
    float x0 = (float)src[d0], x1 = (float)src[d1];
    float x2 = (float)src[d2], x3 = (float)src[d3];

    if (slot >= 24) {  // V: plain transposed copy
        const int hk = slot - 24;
        bf16* vb = vT + (size_t)(b * NKV + hk) * DHD * TT;
        vb[(size_t)d0 * TT + t] = (bf16)x0;
        vb[(size_t)d1 * TT + t] = (bf16)x1;
        vb[(size_t)d2 * TT + t] = (bf16)x2;
        vb[(size_t)d3 * TT + t] = (bf16)x3;
        return;
    }

    float ss = x0 * x0 + x1 * x1 + x2 * x2 + x3 * x3;
#pragma unroll
    for (int msk = 1; msk < 32; msk <<= 1) ss += __shfl_xor(ss, msk, 32);
    const float rn = rsqrtf(ss * (1.0f / 128.0f) + 1e-6f);
    const float* wgt = (slot < 16) ? qw : kw;
    x0 *= rn * (1.0f + wgt[d0]);
    x1 *= rn * (1.0f + wgt[d1]);
    x2 *= rn * (1.0f + wgt[d2]);
    x3 *= rn * (1.0f + wgt[d3]);

    // RoPE: angle(d) = t * theta^(-d/64), pairs (d, d+64) share angle.
    const float LT = 9.210340371976184f / 64.0f;  // ln(10000)/64
    const float a0 = (float)t * __expf(-(float)d0 * LT);
    const float a1 = (float)t * __expf(-(float)d1 * LT);
    float s0, c0, s1, c1;
    __sincosf(a0, &s0, &c0);
    __sincosf(a1, &s1, &c1);
    float o0 = x0 * c0 - x2 * s0, o2 = x2 * c0 + x0 * s0;
    float o1 = x1 * c1 - x3 * s1, o3 = x3 * c1 + x1 * s1;

    if (slot < 16) {
        const float sc = 0.08838834764831845f;  // 1/sqrt(128)
        o0 *= sc; o1 *= sc; o2 *= sc; o3 *= sc;
        bf16* dst = qb + ((size_t)(b * NH + slot) * TT + t) * DHD;
        dst[d0] = (bf16)o0; dst[d1] = (bf16)o1;
        dst[d2] = (bf16)o2; dst[d3] = (bf16)o3;
    } else {
        const int hk = slot - 16;
        bf16* dst = kb + ((size_t)(b * NKV + hk) * TT + t) * DHD;
        dst[d0] = (bf16)o0; dst[d1] = (bf16)o1;
        dst[d2] = (bf16)o2; dst[d3] = (bf16)o3;
    }
}

// ---------------------------------------------------------------------------
// Kernel 3: flash attention, sliding window [t-512, t].
// One wave per 16-query tile; 8 waves per block. Per 32-key chunk:
// 8 WMMA for S = Q*K^T, online softmax in C-layout, P relayout via LDS,
// 8 WMMA for ctx += P*V (V pre-transposed [d][t]).
// ---------------------------------------------------------------------------
__global__ __launch_bounds__(256) void attn_swa(const bf16* __restrict__ q,
                                                const bf16* __restrict__ k,
                                                const bf16* __restrict__ vT,
                                                bf16* __restrict__ cb) {
    __shared__ __align__(16) bf16 lsP[8][16 * 32];

    const int wv   = threadIdx.x >> 5;
    const int lane = threadIdx.x & 31;
    const int half = lane >> 4;
    const int col  = lane & 15;
    const int t0   = (blockIdx.x * 8 + wv) * 16;
    const int bh   = blockIdx.y;
    const int b    = bh >> 4;
    const int h    = bh & 15;
    const int hk   = h >> 1;   // G = 2

    // Q fragments (A-layout 16x32 per 32-wide d-chunk), resident for the loop.
    const bf16* qp = q + ((size_t)(b * NH + h) * TT + t0) * DHD;
    v16bf qf[4];
#pragma unroll
    for (int dc = 0; dc < 4; ++dc) {
        const bf16* p = qp + (size_t)col * DHD + dc * 32 + half * 16;
        qf[dc] = frag16(p, p + 8);
    }

    const bf16* kp = k  + (size_t)(b * NKV + hk) * TT * DHD;
    const bf16* vp = vT + (size_t)(b * NKV + hk) * DHD * TT;

    v8f ctx[8];
#pragma unroll
    for (int j = 0; j < 8; ++j) ctx[j] = zero8();
    float mrow[8], lrow[8];
#pragma unroll
    for (int r = 0; r < 8; ++r) { mrow[r] = -1e30f; lrow[r] = 0.0f; }

    const int ks = (t0 >= WND) ? ((t0 - WND) & ~31) : 0;
    for (int kc = ks; kc < t0 + 16; kc += 32) {
        // ---- S = Q * K^T over this 32-key chunk (two 16-key column tiles)
        v8f s0 = zero8(), s1 = zero8();
        const int kr0 = kc + col, kr1 = kr0 + 16;           // true key ids
        const int a0r = imin(kr0, TT - 1), a1r = imin(kr1, TT - 1);  // clamped
#pragma unroll
        for (int dc = 0; dc < 4; ++dc) {
            const bf16* p0 = kp + (size_t)a0r * DHD + dc * 32 + half * 16;
            const bf16* p1 = kp + (size_t)a1r * DHD + dc * 32 + half * 16;
            v16bf k0 = frag16(p0, p0 + 8);
            v16bf k1 = frag16(p1, p1 + 8);
            s0 = wmma_bf16(qf[dc], k0, s0);
            s1 = wmma_bf16(qf[dc], k1, s1);
        }

        // ---- mask + online softmax (C-layout: lane holds col, 8 rows)
        float cmax[8];
#pragma unroll
        for (int r = 0; r < 8; ++r) {
            const int tq = t0 + half * 8 + r;
            const bool v0 = (kr0 <= tq) && (tq - kr0 <= WND);
            const bool v1 = (kr1 <= tq) && (tq - kr1 <= WND);
            float a = v0 ? s0[r] : -1e30f;
            float bb = v1 ? s1[r] : -1e30f;
            s0[r] = a; s1[r] = bb;
            cmax[r] = fmaxf(a, bb);
        }
#pragma unroll
        for (int r = 0; r < 8; ++r)
#pragma unroll
            for (int msk = 1; msk < 16; msk <<= 1)
                cmax[r] = fmaxf(cmax[r], __shfl_xor(cmax[r], msk, 32));

        float scl[8], psum[8];
#pragma unroll
        for (int r = 0; r < 8; ++r) {
            const float mn = fmaxf(mrow[r], cmax[r]);
            scl[r] = __expf(mrow[r] - mn);
            mrow[r] = mn;
            const float p0 = (s0[r] > -1e29f) ? __expf(s0[r] - mn) : 0.0f;
            const float p1 = (s1[r] > -1e29f) ? __expf(s1[r] - mn) : 0.0f;
            const int mr = half * 8 + r;
            lsP[wv][mr * 32 + col]      = (bf16)p0;
            lsP[wv][mr * 32 + 16 + col] = (bf16)p1;
            psum[r] = p0 + p1;
        }
#pragma unroll
        for (int r = 0; r < 8; ++r)
#pragma unroll
            for (int msk = 1; msk < 16; msk <<= 1)
                psum[r] += __shfl_xor(psum[r], msk, 32);
#pragma unroll
        for (int r = 0; r < 8; ++r) lrow[r] = lrow[r] * scl[r] + psum[r];
#pragma unroll
        for (int j = 0; j < 8; ++j)
#pragma unroll
            for (int r = 0; r < 8; ++r) ctx[j][r] *= scl[r];

        // Same-wave DS ops complete in order; just stop compiler reordering.
        asm volatile("" ::: "memory");

        // ---- P (A-layout 16x32) and ctx += P * V
        const bf16* pp = &lsP[wv][col * 32 + half * 16];
        v16bf pf = frag16(pp, pp + 8);
        const int vk = imin(kc + half * 16, TT - 16);  // clamped key base
#pragma unroll
        for (int j = 0; j < 8; ++j) {
            const bf16* vb = vp + (size_t)(j * 16 + col) * TT + vk;
            v16bf vf = frag16(vb, vb + 8);
            ctx[j] = wmma_bf16(pf, vf, ctx[j]);
        }
    }

    // ---- epilogue: ctx / l -> cb[B*T][H*DH]
    float inv[8];
#pragma unroll
    for (int r = 0; r < 8; ++r) inv[r] = 1.0f / lrow[r];
#pragma unroll
    for (int j = 0; j < 8; ++j)
#pragma unroll
        for (int r = 0; r < 8; ++r) {
            const int row = b * TT + t0 + half * 8 + r;
            cb[(size_t)row * (NH * DHD) + h * DHD + j * 16 + col] =
                (bf16)(ctx[j][r] * inv[r]);
        }
}

// ---------------------------------------------------------------------------
// Launcher
// ---------------------------------------------------------------------------
extern "C" void kernel_launch(void* const* d_in, const int* in_sizes, int n_in,
                              void* d_out, int out_size, void* d_ws, size_t ws_size,
                              hipStream_t stream) {
    (void)in_sizes; (void)n_in; (void)out_size; (void)ws_size;
    const float* x   = (const float*)d_in[0];
    const float* wq  = (const float*)d_in[1];
    const float* wk  = (const float*)d_in[2];
    const float* wv  = (const float*)d_in[3];
    const float* wo  = (const float*)d_in[4];
    const float* qnw = (const float*)d_in[5];
    const float* knw = (const float*)d_in[6];
    float* out = (float*)d_out;

    char* w = (char*)d_ws;
    size_t o = 0;
    bf16* xbf  = (bf16*)(w + o); o += (size_t)4096 * 2048 * 2;  // x bf16
    bf16* wqkv = (bf16*)(w + o); o += (size_t)4096 * 2048 * 2;  // [wq;wk;wv]
    bf16* wobf = (bf16*)(w + o); o += (size_t)2048 * 2048 * 2;  // wo bf16
    bf16* qkvb = (bf16*)(w + o); o += (size_t)4096 * 4096 * 2;  // qkv proj
    bf16* qb   = (bf16*)(w + o); o += (size_t)BN * NH  * TT * DHD * 2;
    bf16* kb   = (bf16*)(w + o); o += (size_t)BN * NKV * TT * DHD * 2;
    bf16* vT   = (bf16*)(w + o); o += (size_t)BN * NKV * DHD * TT * 2;
    bf16* cb   = (bf16*)(w + o); o += (size_t)4096 * 2048 * 2;  // attn ctx

    cvt_f32_bf16<<<8192, 256, 0, stream>>>(x,  xbf, 8388608);
    cvt_f32_bf16<<<4096, 256, 0, stream>>>(wq, wqkv, 4194304);
    cvt_f32_bf16<<<2048, 256, 0, stream>>>(wk, wqkv + (size_t)2048 * 2048, 2097152);
    cvt_f32_bf16<<<2048, 256, 0, stream>>>(wv, wqkv + (size_t)3072 * 2048, 2097152);
    cvt_f32_bf16<<<4096, 256, 0, stream>>>(wo, wobf, 4194304);

    // QKV projection: [4096 x 4096] = x[4096 x 2048] * wqkv^T
    gemm_bf16nt<true><<<dim3(32, 32), 256, 0, stream>>>(xbf, wqkv, qkvb,
                                                        4096, 4096, 2048);
    // RMSNorm + RoPE + scale; V transposed
    rmsrope<<<16384, 256, 0, stream>>>(qkvb, qnw, knw, qb, kb, vT);
    // Sliding-window flash attention
    attn_swa<<<dim3(TT / 128, BN * NH), 256, 0, stream>>>(qb, kb, vT, cb);
    // Output projection: out[4096 x 2048] = cb[4096 x 2048] * wo^T (f32 out)
    gemm_bf16nt<false><<<dim3(16, 32), 256, 0, stream>>>(cb, wobf, out,
                                                         4096, 2048, 2048);
}
